// TransformerEncoderModel_60885456388929
// MI455X (gfx1250) — compile-verified, matrix operand order checked by reading, not run on hardware
//
#include <hip/hip_runtime.h>
#include <math.h>

// ---- problem constants (from reference) ----
#define BATCH   16
#define NSEQ    2048
#define MTOK    (BATCH*NSEQ)   // 32768 token rows
#define DIMC    64
#define FFN     256
#define NHEAD   8
#define DHEAD   8
#define NLOCAL  4
#define WINSZ   128
#define NWIN    (NSEQ/WINSZ)   // 16
#define GOUTC   512

typedef __attribute__((ext_vector_type(16))) _Float16 v16h;
typedef __attribute__((ext_vector_type(2)))  _Float16 v2h;
typedef __attribute__((ext_vector_type(8)))  float    v8f;

__device__ __forceinline__ float warp_sum(float v){
  #pragma unroll
  for (int m=16;m>0;m>>=1) v += __shfl_xor(v, m, 32);
  return v;
}
__device__ __forceinline__ float warp_max(float v){
  #pragma unroll
  for (int m=16;m>0;m>>=1) v = fmaxf(v, __shfl_xor(v, m, 32));
  return v;
}

// ---------------- LayerNorm: one row (DIM=64) per wave32 ----------------
__global__ __launch_bounds__(256)
void k_layernorm(const float* __restrict__ x, const float* __restrict__ g,
                 const float* __restrict__ b, float* __restrict__ y){
  int wave = threadIdx.x >> 5, lane = threadIdx.x & 31;
  int row  = blockIdx.x * 8 + wave;
  const float* xr = x + (size_t)row * DIMC;
  float a0 = xr[lane], a1 = xr[lane + 32];
  float mu  = warp_sum(a0 + a1) * (1.0f / DIMC);
  float d0 = a0 - mu, d1 = a1 - mu;
  float var = warp_sum(d0*d0 + d1*d1) * (1.0f / DIMC);
  float r = rsqrtf(var + 1e-5f);
  y[(size_t)row*DIMC + lane]      = d0*r*g[lane]      + b[lane];
  y[(size_t)row*DIMC + lane + 32] = d1*r*g[lane + 32] + b[lane + 32];
}

// ---------------- Fused WMMA GEMM  Y = act(X@W + bias) (+res) ----------------
// X:[M,K] f32, W:[K,N_] f32 row-major. K%64==0. Compile-time N_ in {64,128,256}.
// grid.x = M/64, block = 256 (8 waves). ACT_: 0 none, 1 gelu, 2 elu, 3 sigmoid.
//
// LDS holds operands pre-swizzled into WMMA 16x16x32 f16 fragment order:
// region(tile, ks) = 32 lanes x 16 halves (1024B); lane reads its whole v16h
// operand as 32 contiguous bytes (2x ds_load_b128). Inverse map (ISA 7.12.2):
//   k5 = k & 31; half = (k5>>3)&1; j = (k5&7) + ((k5>>4)<<3); lane = 16*half + (row&15)
template<int N_, int ACT_>
__global__ __launch_bounds__(256)
void k_gemm_wmma(const float* __restrict__ X, const float* __restrict__ W,
                 const float* __restrict__ bias, const float* __restrict__ res,
                 float* __restrict__ Y, int K){
  constexpr int NUMNT = N_ / 16;
  constexpr int TPW   = NUMNT / 2;            // tiles per wave (compile-time!)
  __shared__ __align__(32) _Float16 Xs[4 * 2 * 512];       // 4 mt x 2 ks x 512 halves
  __shared__ __align__(32) _Float16 Ws[NUMNT * 2 * 512];   // numNT x 2 ks x 512 halves

  const int tid  = threadIdx.x;
  const int wave = tid >> 5, lane = tid & 31;
  const int rowBase = blockIdx.x * 64;
  const int mt  = wave & 3;                   // this wave's 16-row subtile
  const int nt0 = wave >> 2;                  // N-tile parity

  v8f acc[TPW] = {};

  for (int kc = 0; kc < K; kc += 64){
    __syncthreads();
    // ---- stage X chunk (64 rows x 64 k) into fragment order, 2 halves/store ----
    #pragma unroll
    for (int i = tid; i < 64*32; i += 256){
      int r  = i >> 5;
      int c2 = (i & 31) << 1;                 // even k within chunk
      const float* xp = X + (size_t)(rowBase + r)*K + kc + c2;
      float f0 = xp[0], f1 = xp[1];
      int msub = r >> 4, ks = c2 >> 5, k5 = c2 & 31;
      int half = (k5 >> 3) & 1;
      int j    = (k5 & 7) + ((k5 >> 4) << 3);
      int ln   = (half << 4) + (r & 15);
      int offH = ((msub*2 + ks) << 9) + (ln << 4) + j;
      v2h p; p[0] = (_Float16)f0; p[1] = (_Float16)f1;
      *(v2h*)&Xs[offH] = p;
    }
    // ---- stage W chunk (64 k x N_) into fragment order ----
    #pragma unroll
    for (int i = tid; i < 32*N_; i += 256){
      int n  = i & (N_ - 1);
      int k2 = (i / N_) << 1;
      float f0 = W[(size_t)(kc + k2    )*N_ + n];
      float f1 = W[(size_t)(kc + k2 + 1)*N_ + n];
      int nsub = n >> 4, ks = k2 >> 5, k5 = k2 & 31;
      int half = (k5 >> 3) & 1;
      int j    = (k5 & 7) + ((k5 >> 4) << 3);
      int ln   = (half << 4) + (n & 15);
      int offH = ((nsub*2 + ks) << 9) + (ln << 4) + j;
      v2h p; p[0] = (_Float16)f0; p[1] = (_Float16)f1;
      *(v2h*)&Ws[offH] = p;
    }
    // prefetch next K-chunk while this one computes (global_prefetch_b8)
    if (kc + 64 < K){
      __builtin_prefetch(&X[(size_t)(rowBase + (tid >> 2))*K + kc + 64 + ((tid & 3) << 4)], 0, 1);
      __builtin_prefetch(&W[(size_t)(kc + 64 + (tid & 63))*N_], 0, 1);
    }
    __syncthreads();

    #pragma unroll
    for (int ks = 0; ks < 2; ks++){
      v16h a = *(const v16h*)&Xs[((mt*2 + ks) << 9) + (lane << 4)];
      #pragma unroll
      for (int ti = 0; ti < TPW; ti++){
        int nt = nt0 + 2*ti;
        v16h bf = *(const v16h*)&Ws[((nt*2 + ks) << 9) + (lane << 4)];
        acc[ti] = __builtin_amdgcn_wmma_f32_16x16x32_f16(
                    false, a, false, bf, (short)0, acc[ti], false, false);
      }
    }
  }

  // epilogue: D(m,n) -> lane = (m>=8)*16 + n, vgpr = m%8
  const int l15  = lane & 15;
  const int half = lane >> 4;
  #pragma unroll
  for (int ti = 0; ti < TPW; ti++){
    int nt  = nt0 + 2*ti;
    int col = nt*16 + l15;
    int rb  = rowBase + mt*16 + half*8;
    float bb = bias ? bias[col] : 0.0f;
    #pragma unroll
    for (int r = 0; r < 8; r++){
      float v = acc[ti][r] + bb;
      if      (ACT_ == 1) v = 0.5f * v * (1.0f + erff(v * 0.70710678f));  // exact gelu
      else if (ACT_ == 2) v = v > 0.0f ? v : (__expf(v) - 1.0f);          // elu
      else if (ACT_ == 3) v = 1.0f / (1.0f + __expf(-v));                 // sigmoid
      size_t idx = (size_t)(rb + r)*N_ + col;
      if (res) v += res[idx];
      Y[idx] = v;
    }
  }
}

// ---------------- Local windowed attention (heads 0..3, rotary, DH=8) -------
// grid = B*NLOCAL*NWIN blocks, 128 threads (one q row each)
__global__ __launch_bounds__(128)
void k_local_attn(const float* __restrict__ q, const float* __restrict__ k,
                  const float* __restrict__ v, float* __restrict__ out){
  __shared__ float Ks[3*WINSZ][DHEAD];
  __shared__ float Vs[3*WINSZ][DHEAD];
  int bid = blockIdx.x;
  int w = bid % NWIN;
  int h = (bid / NWIN) % NLOCAL;
  int b = bid / (NWIN * NLOCAL);
  const int tid = threadIdx.x;
  const float invf[4] = {1.0f, 0.1f, 0.01f, 0.001f};  // 10000^{-2i/8}, 10000=10^4

  for (int i = tid; i < 3*WINSZ; i += 128){
    int gn = (w - 1)*WINSZ + i;
    if (gn >= 0 && gn < NSEQ){
      const float* kr = k + ((size_t)(b*NSEQ + gn))*DIMC + h*DHEAD;
      const float* vr = v + ((size_t)(b*NSEQ + gn))*DIMC + h*DHEAD;
      float kk[8];
      #pragma unroll
      for (int d = 0; d < 8; d++) kk[d] = kr[d];
      #pragma unroll
      for (int d = 0; d < 4; d++){
        float f = (float)gn * invf[d];
        float c = __cosf(f), s = __sinf(f);
        Ks[i][d]   = kk[d]*c   - kk[d+4]*s;
        Ks[i][d+4] = kk[d+4]*c + kk[d]*s;
      }
      #pragma unroll
      for (int d = 0; d < 8; d++) Vs[i][d] = vr[d];
    }
  }
  __syncthreads();

  int gq = w*WINSZ + tid;
  const float* qr = q + ((size_t)(b*NSEQ + gq))*DIMC + h*DHEAD;
  float qv[8];
  #pragma unroll
  for (int d = 0; d < 8; d++) qv[d] = qr[d];
  #pragma unroll
  for (int d = 0; d < 4; d++){
    float f = (float)gq * invf[d];
    float c = __cosf(f), s = __sinf(f);
    float x1 = qv[d], x2 = qv[d+4];
    qv[d] = x1*c - x2*s;  qv[d+4] = x2*c + x1*s;
  }
  #pragma unroll
  for (int d = 0; d < 8; d++) qv[d] *= 0.35355339f;   // DH^-0.5

  float m = -1e30f, l = 0.0f;
  float o[8] = {0,0,0,0,0,0,0,0};
  int j0 = (w == 0)        ? WINSZ   : 0;
  int j1 = (w == NWIN - 1) ? 2*WINSZ : 3*WINSZ;
  for (int j = j0; j < j1; j++){
    float s = 0.f;
    #pragma unroll
    for (int d = 0; d < 8; d++) s += qv[d] * Ks[j][d];
    float mn = fmaxf(m, s);
    float corr = __expf(m - mn);
    float p    = __expf(s - mn);
    l = l*corr + p;
    #pragma unroll
    for (int d = 0; d < 8; d++) o[d] = o[d]*corr + p*Vs[j][d];
    m = mn;
  }
  float inv = 1.0f / l;
  float* op = out + ((size_t)(b*NSEQ + gq))*DIMC + h*DHEAD;
  #pragma unroll
  for (int d = 0; d < 8; d++) op[d] = o[d]*inv;
}

// ---------------- Linear attention (heads 4..7) ------------------------------
// grid = B*4 blocks (one (b,h)), 256 threads.
__global__ __launch_bounds__(256)
void k_linear_attn(const float* __restrict__ q, const float* __restrict__ k,
                   const float* __restrict__ v, float* __restrict__ out){
  __shared__ float wred[8*8];
  __shared__ float smax[8];
  __shared__ float sctx[64];
  __shared__ float ssum[8];
  int b = blockIdx.x >> 2;
  int h = NLOCAL + (blockIdx.x & 3);
  int tid = threadIdx.x, wave = tid >> 5, lane = tid & 31;

  // pass 1: per-column (d) max of K over n
  float cmax[8];
  #pragma unroll
  for (int d = 0; d < 8; d++) cmax[d] = -1e30f;
  for (int n = tid; n < NSEQ; n += 256){
    const float* kr = k + ((size_t)(b*NSEQ + n))*DIMC + h*DHEAD;
    #pragma unroll
    for (int d = 0; d < 8; d++) cmax[d] = fmaxf(cmax[d], kr[d]);
  }
  #pragma unroll
  for (int d = 0; d < 8; d++) cmax[d] = warp_max(cmax[d]);
  if (lane == 0){
    #pragma unroll
    for (int d = 0; d < 8; d++) wred[wave*8 + d] = cmax[d];
  }
  __syncthreads();
  if (tid < 8){
    float mm = -1e30f;
    for (int wv = 0; wv < 8; wv++) mm = fmaxf(mm, wred[wv*8 + tid]);
    smax[tid] = mm;
  }
  if (tid >= 8  && tid < 72) sctx[tid - 8]  = 0.0f;
  if (tid >= 72 && tid < 80) ssum[tid - 72] = 0.0f;
  __syncthreads();

  // pass 2: unnormalized ctx[d][e] = sum_n exp(k-max) * v, plus column sums
  float ctx[64];
  #pragma unroll
  for (int i = 0; i < 64; i++) ctx[i] = 0.0f;
  float csum[8];
  #pragma unroll
  for (int d = 0; d < 8; d++) csum[d] = 0.0f;
  for (int n = tid; n < NSEQ; n += 256){
    const float* kr = k + ((size_t)(b*NSEQ + n))*DIMC + h*DHEAD;
    const float* vr = v + ((size_t)(b*NSEQ + n))*DIMC + h*DHEAD;
    float p[8], vv[8];
    #pragma unroll
    for (int d = 0; d < 8; d++){ p[d] = __expf(kr[d] - smax[d]); csum[d] += p[d]; }
    #pragma unroll
    for (int e = 0; e < 8; e++) vv[e] = vr[e];
    #pragma unroll
    for (int d = 0; d < 8; d++)
      #pragma unroll
      for (int e = 0; e < 8; e++) ctx[d*8 + e] += p[d]*vv[e];
  }
  for (int i = 0; i < 64; i++){
    float s = warp_sum(ctx[i]);
    if (lane == 0) atomicAdd(&sctx[i], s);     // ds_add_f32
  }
  #pragma unroll
  for (int d = 0; d < 8; d++){
    float s = warp_sum(csum[d]);
    if (lane == 0) atomicAdd(&ssum[d], s);
  }
  __syncthreads();
  if (tid < 64) sctx[tid] = sctx[tid] / ssum[tid >> 3];
  __syncthreads();

  // pass 3: out = softmax(q)/sqrt(d) @ ctx
  for (int n = tid; n < NSEQ; n += 256){
    const float* qr = q + ((size_t)(b*NSEQ + n))*DIMC + h*DHEAD;
    float qv[8];
    float qm = -1e30f;
    #pragma unroll
    for (int d = 0; d < 8; d++){ qv[d] = qr[d]; qm = fmaxf(qm, qv[d]); }
    float qs = 0.f;
    #pragma unroll
    for (int d = 0; d < 8; d++){ qv[d] = __expf(qv[d] - qm); qs += qv[d]; }
    float invq = 0.35355339f / qs;
    float* op = out + ((size_t)(b*NSEQ + n))*DIMC + h*DHEAD;
    #pragma unroll
    for (int e = 0; e < 8; e++){
      float s = 0.f;
      #pragma unroll
      for (int d = 0; d < 8; d++) s += qv[d] * sctx[d*8 + e];
      op[e] = s * invq;
    }
  }
}

// ---------------- small utility kernels (float4-vectorized) ----------------
__global__ void k_split_src(const float4* __restrict__ src, float4* __restrict__ xc,
                            float4* __restrict__ xt){
  size_t i = (size_t)blockIdx.x * blockDim.x + threadIdx.x;  // over MTOK*16
  size_t m = i >> 4; int c = (int)(i & 15);
  xc[i] = src[m*32 + c];
  xt[i] = src[m*32 + 16 + c];
}
__global__ void k_pack_cat(const float4* __restrict__ c, const float4* __restrict__ t,
                           float4* __restrict__ cat){
  size_t i = (size_t)blockIdx.x * blockDim.x + threadIdx.x;  // over MTOK*16
  size_t m = i >> 4; int j = (int)(i & 15);
  cat[m*32 + j]      = c[i];
  cat[m*32 + 16 + j] = t[i];
}
__global__ void k_fuse_mean(const float* __restrict__ c, const float* __restrict__ t,
                            const float* __restrict__ w, float* __restrict__ fmean){
  int b = blockIdx.x; int col = threadIdx.x;   // 128 threads
  float s = 0.f;
  for (int n = 0; n < NSEQ; n++){
    size_t row = (size_t)b*NSEQ + n;
    if (col < 64){ float wv = w[row*64 + col];      s += c[row*64 + col]      * wv; }
    else         { float wv = w[row*64 + col - 64]; s += t[row*64 + col - 64] * (1.0f - wv); }
  }
  fmean[b*128 + col] = s * (1.0f / NSEQ);
}
__global__ void k_out_proj(const float* __restrict__ fmean, const float* __restrict__ Wo,
                           const float* __restrict__ bo, float* __restrict__ outp){
  int b = blockIdx.x; int o = blockIdx.y*256 + threadIdx.x;
  float s = bo[o];
  #pragma unroll 8
  for (int kk = 0; kk < 128; kk++) s += fmean[b*128 + kk] * Wo[kk*GOUTC + o];
  outp[(size_t)b*GOUTC + o] = s;
}

// ---------------- host-side orchestration ----------------
extern "C" void kernel_launch(void* const* d_in, const int* in_sizes, int n_in,
                              void* d_out, int out_size, void* d_ws, size_t ws_size,
                              hipStream_t stream){
  (void)in_sizes; (void)n_in; (void)out_size; (void)ws_size;
  const float* src = (const float*)d_in[0];
  auto P = [&](int i){ return (const float*)d_in[i]; };
  // layer leaves (insertion order): 0 ln1_g 1 ln1_b 2 wq 3 wk 4 wv 5 wo 6 bo
  //                                 7 ln2_g 8 ln2_b 9 w1 10 b1 11 w2 12 b2
  float* ws = (float*)d_ws;
  const size_t M64 = (size_t)MTOK * 64;
  float* xc  = ws;  ws += M64;
  float* xt  = ws;  ws += M64;
  float* tmp = ws;  ws += M64;
  float* qb  = ws;  ws += M64;
  float* kb  = ws;  ws += M64;
  float* vb  = ws;  ws += M64;
  float* ab  = ws;  ws += M64;
  float* h1  = ws;  ws += (size_t)MTOK * 256;   // also reused as 'cat' [M,128]
  float* fmean = ws;

  k_split_src<<<(int)(MTOK*16/256), 256, 0, stream>>>((const float4*)src,
                                                      (float4*)xc, (float4*)xt);

  for (int enc = 0; enc < 2; enc++){
    float* x = enc ? xt : xc;
    int base0 = 1 + enc*65;
    for (int L = 0; L < 5; L++){
      int p = base0 + L*13;
      k_layernorm<<<MTOK/8, 256, 0, stream>>>(x, P(p+0), P(p+1), tmp);
      k_gemm_wmma<64,0><<<MTOK/64, 256, 0, stream>>>(tmp, P(p+2), nullptr, nullptr, qb, 64);
      k_gemm_wmma<64,0><<<MTOK/64, 256, 0, stream>>>(tmp, P(p+3), nullptr, nullptr, kb, 64);
      k_gemm_wmma<64,0><<<MTOK/64, 256, 0, stream>>>(tmp, P(p+4), nullptr, nullptr, vb, 64);
      k_local_attn<<<BATCH*NLOCAL*NWIN, 128, 0, stream>>>(qb, kb, vb, ab);
      k_linear_attn<<<BATCH*4, 256, 0, stream>>>(qb, kb, vb, ab);
      k_gemm_wmma<64,0><<<MTOK/64, 256, 0, stream>>>(ab, P(p+5), P(p+6), x, x, 64);
      k_layernorm<<<MTOK/8, 256, 0, stream>>>(x, P(p+7), P(p+8), tmp);
      k_gemm_wmma<256,1><<<MTOK/64, 256, 0, stream>>>(tmp, P(p+9), P(p+10), nullptr, h1, 64);
      k_gemm_wmma<64,0><<<MTOK/64, 256, 0, stream>>>(h1, P(p+11), P(p+12), x, x, 256);
    }
  }

  // gating head: w = sigmoid(elu(cat@W0+b0)@W1+b1); output order: f_global, then w
  float* wgate = (float*)d_out + (size_t)BATCH*GOUTC;
  k_pack_cat<<<(int)(MTOK*16/256), 256, 0, stream>>>((const float4*)xc, (const float4*)xt,
                                                     (float4*)h1);
  k_gemm_wmma<64,2><<<MTOK/64, 256, 0, stream>>>(h1,  P(131), P(132), nullptr, tmp,   128);
  k_gemm_wmma<64,3><<<MTOK/64, 256, 0, stream>>>(tmp, P(133), P(134), nullptr, wgate,  64);
  k_fuse_mean<<<BATCH, 128, 0, stream>>>(xc, xt, wgate, fmean);
  k_out_proj<<<dim3(BATCH, 2), 256, 0, stream>>>(fmean, P(135), P(136), (float*)d_out);
}
